// SelfAttention_43215960932520
// MI455X (gfx1250) — matrix operand order repaired and checked by reading, not compiled, not run on hardware
//
#include <hip/hip_runtime.h>

// Problem constants (match reference)
#define N_TOK 2048
#define CXC   768
#define CSC   384
#define NH    16
#define HD    48

typedef __attribute__((ext_vector_type(16))) __bf16 v16bf;
typedef __attribute__((ext_vector_type(8)))  __bf16 v8bf;
typedef __attribute__((ext_vector_type(8)))  float  v8f;

__device__ __forceinline__ float sigmoidf_(float x){ return 1.0f/(1.0f+__expf(-x)); }

// ---- CDNA5 async global->LDS copies (ASYNCcnt-tracked, no VGPR round trip) ----
__device__ __forceinline__ void async_ld_b128(void* lds, const void* g){
  unsigned l = (unsigned)(unsigned long long)(uintptr_t)lds;   // LDS byte offset (addr[31:0])
  asm volatile("global_load_async_to_lds_b128 %0, %1, off"
               :: "v"(l), "v"((unsigned long long)(uintptr_t)g) : "memory");
}
__device__ __forceinline__ void async_ld_b64(void* lds, const void* g){
  unsigned l = (unsigned)(unsigned long long)(uintptr_t)lds;
  asm volatile("global_load_async_to_lds_b64 %0, %1, off"
               :: "v"(l), "v"((unsigned long long)(uintptr_t)g) : "memory");
}
__device__ __forceinline__ void wait_async(){
  asm volatile("s_wait_asynccnt 0x0" ::: "memory");
}

// ---- WMMA fragment loaders (CDNA5 ISA 7.12.2 layouts, bf16 16x16x32) ----
// A (16x32): lane m = lane&15; lanes 0-15 hold K 0..7 & 16..23, lanes 16-31 hold 8..15 & 24..31
__device__ __forceinline__ v16bf ld_fragA(const __bf16* rowp, int kh){
  v16bf f;
  v8bf lo = *(const v8bf*)(rowp + 8*kh);
  v8bf hi = *(const v8bf*)(rowp + 16 + 8*kh);
#pragma unroll
  for (int e=0;e<8;++e){ f[e]=lo[e]; f[8+e]=hi[e]; }
  return f;
}
// B (32x16): column n = lane&15 per lane; lanes 0-15 hold K 0..7 & 16..23, lanes 16-31 hold 8..15 & 24..31.
__device__ __forceinline__ v16bf ld_fragB(const __bf16* colp, int kh){
  v16bf f;
  v8bf lo = *(const v8bf*)(colp + 8*kh);
  v8bf hi = *(const v8bf*)(colp + 16 + 8*kh);
#pragma unroll
  for (int e=0;e<8;++e){ f[e]=lo[e]; f[8+e]=hi[e]; }
  return f;
}

// ---------------- f32 -> bf16 transposed weight convert ----------------
__global__ void cvt_t_bf16(const float* __restrict__ s, __bf16* __restrict__ d, int K){
  int total = K * CXC;
  for (int i = blockIdx.x*blockDim.x + threadIdx.x; i < total; i += gridDim.x*blockDim.x){
    int k = i / CXC, n = i % CXC;
    d[(size_t)n*K + k] = (__bf16)s[i];
  }
}

// ---------------- fused LayerNorm (x and single_cond) ----------------
__device__ __forceinline__ void block_reduce2(float& a, float& b, float* sh){
#pragma unroll
  for (int off=16; off>0; off>>=1){ a += __shfl_xor(a, off, 32); b += __shfl_xor(b, off, 32); }
  int wid = threadIdx.x >> 5;
  __syncthreads();
  if ((threadIdx.x & 31) == 0){ sh[2*wid] = a; sh[2*wid+1] = b; }
  __syncthreads();
  float ta=0.f, tb=0.f;
#pragma unroll
  for (int i=0;i<8;++i){ ta += sh[2*i]; tb += sh[2*i+1]; }
  a = ta; b = tb;
}

__global__ __launch_bounds__(256)
void ln_kernel(const float* __restrict__ x, const float* __restrict__ scond,
               const float* __restrict__ lnw,
               __bf16* __restrict__ xn, __bf16* __restrict__ sc)
{
  __shared__ float sh[16];
  int r = blockIdx.x;
  const float* xr = x + (size_t)r*CXC;
  float s0=0.f, s1=0.f;
  for (int c=threadIdx.x; c<CXC; c+=256){ float v=xr[c]; s0+=v; s1+=v*v; }
  block_reduce2(s0, s1, sh);
  float mu = s0*(1.0f/CXC);
  float rstd = rsqrtf(fmaxf(s1*(1.0f/CXC)-mu*mu, 0.f)+1e-5f);
  __bf16* xnr = xn + (size_t)r*CXC;
  for (int c=threadIdx.x; c<CXC; c+=256) xnr[c] = (__bf16)((xr[c]-mu)*rstd);

  const float* sr = scond + (size_t)r*CSC;
  s0=0.f; s1=0.f;
  for (int c=threadIdx.x; c<CSC; c+=256){ float v=sr[c]; s0+=v; s1+=v*v; }
  block_reduce2(s0, s1, sh);
  mu = s0*(1.0f/CSC);
  rstd = rsqrtf(fmaxf(s1*(1.0f/CSC)-mu*mu, 0.f)+1e-5f);
  __bf16* scr = sc + (size_t)r*CSC;
  for (int c=threadIdx.x; c<CSC; c+=256) scr[c] = (__bf16)((sr[c]-mu)*rstd*lnw[c]);
}

// ---------------- generic WMMA GEMM: out = act((A @ Wt^T + bias) * oscale) ----------------
// A:  M x K bf16 row-major.  Wt: CXC x K bf16 row-major (pre-transposed weight).
// mode 0: f32 out; 1: sigmoid->f32; 2: bf16 out (scaled); 3: f32 out * aux; 4: bf16 transposed out [n][m]
#define GM 64
#define GN 64
#define GK 64

__global__ __launch_bounds__(256)
void gemm_bf16(const __bf16* __restrict__ A, const __bf16* __restrict__ Wt,
               const float* __restrict__ bias, const float* __restrict__ aux,
               float* __restrict__ outF, __bf16* __restrict__ outB,
               int M, int K, int mode, float oscale)
{
  __shared__ __bf16 lA[2][GM][GK];   // double-buffered [m][k]
  __shared__ __bf16 lB[2][GN][GK];   // double-buffered [n][k]
  int tid = threadIdx.x;
  int m0 = blockIdx.y * GM;
  int n0 = blockIdx.x * GN;
  int wid = tid >> 5, lane = tid & 31;
  int lane15 = lane & 15, kh = lane >> 4;
  int mb = (wid >> 1) * 16;     // 4x2 wave grid over 64x64 tile
  int nb = (wid & 1) * 32;
  v8f acc0 = {}, acc1 = {};
  (void)M;

  int ldr = tid >> 2;           // 0..63 tile row
  int ldc = (tid & 3) * 16;     // 0,16,32,48 within 64-wide K tile

  auto issue_tile = [&](int buf, int k0){
    const __bf16* ap = A + (size_t)(m0+ldr)*K + k0 + ldc;
    async_ld_b128(&lA[buf][ldr][ldc],     ap);
    async_ld_b128(&lA[buf][ldr][ldc + 8], ap + 8);
    const __bf16* bp = Wt + (size_t)(n0+ldr)*K + k0 + ldc;
    async_ld_b128(&lB[buf][ldr][ldc],     bp);
    async_ld_b128(&lB[buf][ldr][ldc + 8], bp + 8);
  };

  int iters = K / GK;
  issue_tile(0, 0);
  for (int it = 0; it < iters; ++it){
    int ib = it & 1;
    wait_async();          // my copies for buffer ib have landed
    __syncthreads();       // everyone's copies have landed
    if (it + 1 < iters) issue_tile(ib ^ 1, (it + 1) * GK);   // prefetch next tile
#pragma unroll
    for (int ks = 0; ks < 2; ++ks){
      v16bf fa  = ld_fragA(&lA[ib][mb + lane15][32*ks], kh);
      v16bf fb0 = ld_fragB(&lB[ib][nb + lane15][32*ks], kh);
      v16bf fb1 = ld_fragB(&lB[ib][nb + 16 + lane15][32*ks], kh);
      acc0 = __builtin_amdgcn_wmma_f32_16x16x32_bf16(false, fa, false, fb0, (short)0, acc0, false, false);
      acc1 = __builtin_amdgcn_wmma_f32_16x16x32_bf16(false, fa, false, fb1, (short)0, acc1, false, false);
    }
  }

  v8f accs[2] = {acc0, acc1};
#pragma unroll
  for (int f=0; f<2; ++f){
    int n = n0 + nb + 16*f + lane15;
    float bv = bias ? bias[n] : 0.0f;
    if (mode == 4){
      __bf16 tmp[8];
#pragma unroll
      for (int v=0; v<8; ++v) tmp[v] = (__bf16)(accs[f][v] + bv);
      *(uint4*)(outB + (size_t)n*N_TOK + m0 + mb + 8*kh) = *(const uint4*)tmp;
      continue;
    }
#pragma unroll
    for (int v=0; v<8; ++v){
      int m = m0 + mb + v + 8*kh;
      size_t idx = (size_t)m*CXC + n;
      float val = accs[f][v] + bv;
      if (mode == 0)      outF[idx] = val;
      else if (mode == 1) outF[idx] = sigmoidf_(val);
      else if (mode == 2) outB[idx] = (__bf16)(val * oscale);
      else                outF[idx] = val * aux[idx];
    }
  }
}

// ---------------- adaLN combine: xa = sigmoid(scale)*xn + bias ----------------
__global__ void xa_kernel(const float* __restrict__ scale_sig, const float* __restrict__ bias_lin,
                          const __bf16* __restrict__ xn, __bf16* __restrict__ xa, int n){
  for (int i = blockIdx.x*blockDim.x + threadIdx.x; i < n; i += gridDim.x*blockDim.x)
    xa[i] = (__bf16)(scale_sig[i]*(float)xn[i] + bias_lin[i]);
}

// ---------------- flash attention with pair bias + mask + fused gating ----------------
// Grid: (N/128 query tiles, NH heads). 8 waves; wave w owns query rows [128*bx + 16w, +16).
// Qm already carries the D^-0.5 scale.  VmT is pre-transposed: [n = h*HD + d][token].
__global__ __launch_bounds__(256)
void attn_kernel(const __bf16* __restrict__ Qm, const __bf16* __restrict__ Km,
                 const __bf16* __restrict__ VmT, const float* __restrict__ pair,
                 const int* __restrict__ mask, const float* __restrict__ gate_sig,
                 __bf16* __restrict__ wa)
{
  __shared__ __bf16 sQ[128][64];        // query rows, d padded 48->64 with zeros
  __shared__ __bf16 sK[2][64][64];      // double-buffered key rows, d padded
  __shared__ __bf16 sV[2][HD][64];      // double-buffered V^T: [d][kk]
  __shared__ __bf16 sP[8][16][64];      // per-wave P staging (C-layout -> A-layout via LDS)

  int tid = threadIdx.x;
  int h  = blockIdx.y;
  int qb = blockIdx.x * 128;
  int wid = tid >> 5, lane = tid & 31, lane15 = lane & 15, kh = lane >> 4;
  const uint2 zero8 = make_uint2(0u, 0u);

  // static zero padding (columns 48..63), done once
  for (int i=tid; i<128*4; i+=256){ int r=i>>2, ch=i&3; *(uint2*)&sQ[r][48+ch*4] = zero8; }
  for (int i=tid; i<64*4; i+=256){
    int r=i>>2, ch=i&3;
    *(uint2*)&sK[0][r][48+ch*4] = zero8;
    *(uint2*)&sK[1][r][48+ch*4] = zero8;
  }

  // async Q tile: 128 rows x 12 chunks of 4 bf16 (8B granules; head stride 96B)
  for (int i=tid; i<128*12; i+=256){
    int r = i/12, ch = i%12;
    async_ld_b64(&sQ[r][ch*4], Qm + (size_t)(qb+r)*CXC + h*HD + ch*4);
  }
  // async K/V tiles for kb = 0 into buffer 0
  for (int i=tid; i<64*12; i+=256){
    int r = i/12, ch = i%12;
    async_ld_b64(&sK[0][r][ch*4], Km + (size_t)(0+r)*CXC + h*HD + ch*4);
  }
  for (int i=tid; i<HD*8; i+=256){
    int d = i>>3, ch = i&7;
    async_ld_b128(&sV[0][d][ch*8], VmT + (size_t)(h*HD + d)*N_TOK + 0 + ch*8);
  }

  float run_m[8], run_l[8];
  v8f o[3] = {};
#pragma unroll
  for (int v=0; v<8; ++v){ run_m[v] = -1e30f; run_l[v] = 0.f; }

  for (int kb=0, it=0; kb<N_TOK; kb+=64, ++it){
    int ib = it & 1;
    wait_async();          // my async copies for buffer ib landed
    __syncthreads();       // all waves' copies landed; prev compute on ib^1 done
    if (kb + 64 < N_TOK){  // prefetch next key block into the other buffer
      int kb2 = kb + 64;
      for (int i=tid; i<64*12; i+=256){
        int r = i/12, ch = i%12;
        async_ld_b64(&sK[ib^1][r][ch*4], Km + (size_t)(kb2+r)*CXC + h*HD + ch*4);
      }
      for (int i=tid; i<HD*8; i+=256){
        int d = i>>3, ch = i&7;
        async_ld_b128(&sV[ib^1][d][ch*8], VmT + (size_t)(h*HD + d)*N_TOK + kb2 + ch*8);
      }
    }

    // S = Q @ K^T over padded d=64 (two 32-K WMMA steps), 64 keys
    v8f sfrag[4] = {};
#pragma unroll
    for (int ks=0; ks<2; ++ks){
      v16bf fa = ld_fragA(&sQ[wid*16 + lane15][32*ks], kh);
#pragma unroll
      for (int f=0; f<4; ++f){
        v16bf fb = ld_fragB(&sK[ib][16*f + lane15][32*ks], kh);
        sfrag[f] = __builtin_amdgcn_wmma_f32_16x16x32_bf16(false, fa, false, fb, (short)0, sfrag[f], false, false);
      }
    }

    // pair bias + key mask (this is the 268MB bandwidth stream)
    const float* prow = pair + ((size_t)h*N_TOK + (size_t)(qb + wid*16 + 8*kh))*N_TOK + kb;
#pragma unroll
    for (int f=0; f<4; ++f){
      int kcol = 16*f + lane15;
      int mk = mask[kb + kcol];
#pragma unroll
      for (int v=0; v<8; ++v){
        float lg = sfrag[f][v] + prow[(size_t)v*N_TOK + kcol];
        sfrag[f][v] = mk ? lg : -1e9f;
      }
      if (kb + 64 < N_TOK) __builtin_prefetch(prow + kcol + 64, 0, 0);
    }

    // online softmax; row m = v + 8*kh, reduce over 16 lanes of the half (N dim)
#pragma unroll
    for (int v=0; v<8; ++v){
      float x = fmaxf(fmaxf(sfrag[0][v], sfrag[1][v]), fmaxf(sfrag[2][v], sfrag[3][v]));
#pragma unroll
      for (int off=1; off<16; off<<=1) x = fmaxf(x, __shfl_xor(x, off, 32));
      float nm = fmaxf(run_m[v], x);
      float corr = __expf(run_m[v] - nm);
      run_m[v] = nm;
      float rs = 0.f;
#pragma unroll
      for (int f=0; f<4; ++f){
        float p = __expf(sfrag[f][v] - nm);
        sfrag[f][v] = p;
        rs += p;
      }
#pragma unroll
      for (int off=1; off<16; off<<=1) rs += __shfl_xor(rs, off, 32);
      run_l[v] = run_l[v]*corr + rs;
      o[0][v] *= corr; o[1][v] *= corr; o[2][v] *= corr;
    }

    // stage P (C layout) into LDS, reload as A fragments
#pragma unroll
    for (int f=0; f<4; ++f)
#pragma unroll
      for (int v=0; v<8; ++v)
        sP[wid][v + 8*kh][16*f + lane15] = (__bf16)sfrag[f][v];

    // O += P @ V  (K = 64 keys, two 32-K steps; N = 48 dims = 3 fragments)
#pragma unroll
    for (int ks=0; ks<2; ++ks){
      v16bf fp = ld_fragA(&sP[wid][lane15][32*ks], kh);
#pragma unroll
      for (int df=0; df<3; ++df){
        v16bf fv = ld_fragB(&sV[ib][16*df + lane15][32*ks], kh);
        o[df] = __builtin_amdgcn_wmma_f32_16x16x32_bf16(false, fp, false, fv, (short)0, o[df], false, false);
      }
    }
  }

  // normalize, gate, write wa (bf16)
  float inv[8];
#pragma unroll
  for (int v=0; v<8; ++v) inv[v] = 1.0f / fmaxf(run_l[v], 1e-20f);
#pragma unroll
  for (int df=0; df<3; ++df)
#pragma unroll
    for (int v=0; v<8; ++v){
      int m = qb + wid*16 + v + 8*kh;
      int c = h*HD + 16*df + lane15;
      size_t idx = (size_t)m*CXC + c;
      float val = o[df][v] * inv[v] * gate_sig[idx];
      wa[idx] = (__bf16)val;
    }
}

// ---------------- host orchestration ----------------
extern "C" void kernel_launch(void* const* d_in, const int* in_sizes, int n_in,
                              void* d_out, int out_size, void* d_ws, size_t ws_size,
                              hipStream_t stream)
{
  (void)in_sizes; (void)n_in; (void)out_size; (void)ws_size;
  const float* x          = (const float*)d_in[0];
  const int*   mask       = (const int*)  d_in[1];
  const float* pair       = (const float*)d_in[2];
  const float* scond      = (const float*)d_in[3];
  const float* ln_sc_w    = (const float*)d_in[4];
  const float* sc_scale_w = (const float*)d_in[5];
  const float* sc_scale_b = (const float*)d_in[6];
  const float* sc_bias_w  = (const float*)d_in[7];
  const float* q_w        = (const float*)d_in[8];
  const float* q_b        = (const float*)d_in[9];
  const float* k_w        = (const float*)d_in[10];
  const float* v_w        = (const float*)d_in[11];
  const float* gate_w     = (const float*)d_in[12];
  const float* out_w      = (const float*)d_in[13];
  const float* azc_w      = (const float*)d_in[14];
  const float* azc_b      = (const float*)d_in[15];
  float* out = (float*)d_out;

  char* ws = (char*)d_ws;
  size_t off = 0;
  auto alloc = [&](size_t bytes)->char*{
    char* p = ws + off;
    off += (bytes + 255) & ~(size_t)255;
    return p;
  };
  // bf16 transposed weights (~7.7 MB)
  __bf16* w_scale = (__bf16*)alloc((size_t)CSC*CXC*2);
  __bf16* w_biasw = (__bf16*)alloc((size_t)CSC*CXC*2);
  __bf16* w_q     = (__bf16*)alloc((size_t)CXC*CXC*2);
  __bf16* w_k     = (__bf16*)alloc((size_t)CXC*CXC*2);
  __bf16* w_v     = (__bf16*)alloc((size_t)CXC*CXC*2);
  __bf16* w_gate  = (__bf16*)alloc((size_t)CXC*CXC*2);
  __bf16* w_out   = (__bf16*)alloc((size_t)CXC*CXC*2);
  __bf16* w_azc   = (__bf16*)alloc((size_t)CSC*CXC*2);
  // activations (~44 MB)
  __bf16* xn        = (__bf16*)alloc((size_t)N_TOK*CXC*2);
  __bf16* scb       = (__bf16*)alloc((size_t)N_TOK*CSC*2);
  float*  scale_sig = (float*) alloc((size_t)N_TOK*CXC*4);
  float*  bias_lin  = (float*) alloc((size_t)N_TOK*CXC*4);
  __bf16* xa        = (__bf16*)alloc((size_t)N_TOK*CXC*2);
  __bf16* qb        = (__bf16*)alloc((size_t)N_TOK*CXC*2);
  __bf16* kbf       = (__bf16*)alloc((size_t)N_TOK*CXC*2);
  __bf16* vT        = (__bf16*)alloc((size_t)N_TOK*CXC*2);   // [n][token]
  float*  gate_sig  = (float*) alloc((size_t)N_TOK*CXC*4);
  float*  azc_sig   = (float*) alloc((size_t)N_TOK*CXC*4);
  __bf16* wabuf     = (__bf16*)alloc((size_t)N_TOK*CXC*2);

  // 1) transposed weight conversion (once per launch)
  cvt_t_bf16<<<512,256,0,stream>>>(sc_scale_w, w_scale, CSC);
  cvt_t_bf16<<<512,256,0,stream>>>(sc_bias_w,  w_biasw, CSC);
  cvt_t_bf16<<<512,256,0,stream>>>(q_w,    w_q,    CXC);
  cvt_t_bf16<<<512,256,0,stream>>>(k_w,    w_k,    CXC);
  cvt_t_bf16<<<512,256,0,stream>>>(v_w,    w_v,    CXC);
  cvt_t_bf16<<<512,256,0,stream>>>(gate_w, w_gate, CXC);
  cvt_t_bf16<<<512,256,0,stream>>>(out_w,  w_out,  CXC);
  cvt_t_bf16<<<512,256,0,stream>>>(azc_w,  w_azc,  CSC);

  // 2) layernorms
  ln_kernel<<<N_TOK,256,0,stream>>>(x, scond, ln_sc_w, xn, scb);

  dim3 gg(CXC/GN, N_TOK/GM);
  const float qscale = 0.14433756729740643f;  // 48^-0.5
  // 3) adaLN: sigmoid(sc@scale_w + b) and sc@bias_w, then combine into xa
  gemm_bf16<<<gg,256,0,stream>>>(scb, w_scale, sc_scale_b, nullptr, scale_sig, nullptr, N_TOK, CSC, 1, 1.f);
  gemm_bf16<<<gg,256,0,stream>>>(scb, w_biasw, nullptr,    nullptr, bias_lin,  nullptr, N_TOK, CSC, 0, 1.f);
  xa_kernel<<<1536,256,0,stream>>>(scale_sig, bias_lin, xn, xa, N_TOK*CXC);

  // 4) projections (q carries D^-0.5; v stored transposed [n][token])
  gemm_bf16<<<gg,256,0,stream>>>(xa, w_q,    q_b,     nullptr, nullptr,  qb,      N_TOK, CXC, 2, qscale);
  gemm_bf16<<<gg,256,0,stream>>>(xa, w_k,    nullptr, nullptr, nullptr,  kbf,     N_TOK, CXC, 2, 1.f);
  gemm_bf16<<<gg,256,0,stream>>>(xa, w_v,    nullptr, nullptr, nullptr,  vT,      N_TOK, CXC, 4, 1.f);
  gemm_bf16<<<gg,256,0,stream>>>(xa, w_gate, nullptr, nullptr, gate_sig, nullptr, N_TOK, CXC, 1, 1.f);
  gemm_bf16<<<gg,256,0,stream>>>(scb, w_azc, azc_b,   nullptr, azc_sig,  nullptr, N_TOK, CSC, 1, 1.f);

  // 5) flash attention with pair bias + mask; gating fused into wa write
  attn_kernel<<<dim3(N_TOK/128, NH),256,0,stream>>>(qb, kbf, vT, pair, mask, gate_sig, wabuf);

  // 6) out = (wa @ out_w) * sigmoid(sc@azc_w + azc_b)   [aux multiply fused]
  gemm_bf16<<<gg,256,0,stream>>>(wabuf, w_out, nullptr, azc_sig, out, nullptr, N_TOK, CXC, 3, 1.f);
}
